// Correlation_31842887533355
// MI455X (gfx1250) — matrix operand order repaired and checked by reading, not compiled, not run on hardware
//
#include <hip/hip_runtime.h>
#include <hip/hip_bf16.h>
#include <stdint.h>

typedef __attribute__((ext_vector_type(16))) __bf16 v16bf;
typedef __attribute__((ext_vector_type(8)))  float  v8f;
typedef __attribute__((ext_vector_type(8)))  __bf16 bf16x8;
typedef __attribute__((ext_vector_type(4)))  __bf16 bf16x4;
typedef __attribute__((ext_vector_type(4)))  float  f32x4;

#define CC   128          // channels (K of the GEMM)
#define HH   96
#define WW   192
#define NOFF 21           // 21 displacements per axis, stride 2, pad 20
#define PITCH 136         // bf16 per LDS row (128 + 8 pad; 272B row keeps 16B alignment)
#define F1_ELEMS (WW * PITCH)          // f1 row cache [x][c]
#define F2_ROWS 288                     // f2 halo cache: xh = x2 + 32, x2 in [-32,255]
#define F2_ELEMS (F2_ROWS * PITCH)
#define PITCH2 196                      // f32 pitch of P-band staging (bank stride 4)
#define POUT_F32 (NOFF * PITCH2)        // 4116 floats
#define LDS_SZ ((F1_ELEMS + F2_ELEMS) * 2 + (POUT_F32 + 32) * 4)  // + 32f trash
#define BLK 384           // 12 waves: (6 t-tiles) x (2 parities)
#define PLANE (HH * WW)   // 18432
#define TENSOR_ELEMS ((size_t)4 * CC * PLANE)          // 9.44M
#define WS_NEED (2 * TENSOR_ELEMS * sizeof(__bf16))    // 37.75 MB

// A fragment (16x32 bf16): lane<16 -> M=lane, K = {8h..8h+7} U {8h+16..8h+23}
static __device__ inline v16bf load_frag_a(const __bf16* f1s, int t0, int par,
                                           int lane, int kc) {
  const int m    = lane & 15;
  const int half = lane >> 4;
  const int x1   = 2 * (t0 + m) + par;
  const int c1   = kc * 32 + 8 * half;
  const __bf16* p = f1s + x1 * PITCH + c1;
  bf16x8 lo = *(const bf16x8*)(p);
  bf16x8 hi = *(const bf16x8*)(p + 16);
  v16bf a;
#pragma unroll
  for (int i = 0; i < 8; ++i) { a[i] = lo[i]; a[i + 8] = hi[i]; }
  return a;
}

// B fragment (32x16 bf16): lane<16 -> N=lane, K = {8h..8h+7} U {8h+16..8h+23}
static __device__ inline v16bf load_frag_b(const __bf16* f2s, int t0, int par,
                                           int lane, int st, int kc) {
  const int n    = lane & 15;
  const int half = lane >> 4;
  const int s    = t0 - 16 + st * 16 + n;
  const int xh   = 2 * s + par + 32;
  const int k0   = kc * 32 + 8 * half;
  const __bf16* p = f2s + xh * PITCH + k0;
  bf16x8 lo = *(const bf16x8*)(p);
  bf16x8 hi = *(const bf16x8*)(p + 16);
  v16bf b;
#pragma unroll
  for (int i = 0; i < 8; ++i) { b[i] = lo[i]; b[i + 8] = hi[i]; }
  return b;
}

// Software-pipelined 12-WMMA band GEMM: fragments for step q+1 are issued
// before the WMMA of step q, so each s_wait_dscnt only covers older loads.
static __device__ inline void band_gemm(const __bf16* f1s, const __bf16* f2s,
                                        int t0, int par, int lane, v8f acc[3]) {
  v16bf a  = load_frag_a(f1s, t0, par, lane, 0);
  v16bf bb = load_frag_b(f2s, t0, par, lane, 0, 0);
#pragma unroll
  for (int q = 0; q < 12; ++q) {
    const int kc = q / 3, st = q - kc * 3;
    const v16bf acur = a;
    const v16bf bcur = bb;
    if (q < 11) {
      const int qn = q + 1;
      const int kcn = qn / 3, stn = qn - kcn * 3;
      if (stn == 0) a = load_frag_a(f1s, t0, par, lane, kcn);
      bb = load_frag_b(f2s, t0, par, lane, stn, kcn);
    }
    acc[st] = __builtin_amdgcn_wmma_f32_16x16x32_bf16(
        false, acur, false, bcur, (short)0, acc[st], false, false);
  }
}

// Branch-free band scatter + coalesced writeback helpers.
static __device__ inline void band_scatter(float* pout, const v8f acc[3],
                                           int t0, int par, int lane) {
  const int half = lane >> 4;
  const int nl   = lane & 15;
#pragma unroll
  for (int st = 0; st < 3; ++st) {
#pragma unroll
    for (int rr = 0; rr < 8; ++rr) {
      const int i   = rr + 8 * half;
      const int dyi = st * 16 + nl - i - 6;
      const int x   = 2 * (t0 + i) + par;
      const int off = ((unsigned)dyi <= 20u) ? (dyi * PITCH2 + x)
                                             : (POUT_F32 + lane);
      pout[off] = acc[st][rr];
    }
  }
}

// ---------------------------------------------------------------------------
// Pass 1: convert f32 NCHW -> bf16 [b][row][x][c] (c contiguous, staging-ready)
// ---------------------------------------------------------------------------
__global__ __launch_bounds__(BLK) void cvt_kernel(
    const float* __restrict__ feat1, const float* __restrict__ feat2,
    __bf16* __restrict__ w) {
  const int row = blockIdx.x, b = blockIdx.y, t = blockIdx.z;
  const float* src = (t ? feat2 : feat1) + (size_t)b * CC * PLANE + (size_t)row * WW;
  __bf16* dst = w + (size_t)t * TENSOR_ELEMS + ((size_t)(b * HH + row) * WW) * CC;

  const int cg0 = threadIdx.x & 7;
  const int x0  = (threadIdx.x >> 3) * 4;
#pragma unroll
  for (int it = 0; it < 4; ++it) {
    const int c0 = cg0 * 4 + it * 32;
    f32x4 v0 = *(const f32x4*)(src + (size_t)(c0 + 0) * PLANE + x0);
    f32x4 v1 = *(const f32x4*)(src + (size_t)(c0 + 1) * PLANE + x0);
    f32x4 v2 = *(const f32x4*)(src + (size_t)(c0 + 2) * PLANE + x0);
    f32x4 v3 = *(const f32x4*)(src + (size_t)(c0 + 3) * PLANE + x0);
#pragma unroll
    for (int xi = 0; xi < 4; ++xi) {
      bf16x4 p;
      p[0] = (__bf16)v0[xi]; p[1] = (__bf16)v1[xi];
      p[2] = (__bf16)v2[xi]; p[3] = (__bf16)v3[xi];
      *(bf16x4*)(dst + (size_t)(x0 + xi) * CC + c0) = p;
    }
  }
}

// Issue one 16B async global->LDS transfer (GVS mode: SGPR base + VGPR offset).
static __device__ inline void async_copy16(uint32_t lds_addr, uint32_t goff,
                                           const __bf16* base) {
  asm volatile("global_load_async_to_lds_b128 %0, %1, %2"
               :: "v"(lds_addr), "v"(goff), "s"(base) : "memory");
}
static __device__ inline void async_wait0() {
  asm volatile("s_wait_asynccnt 0x0" ::: "memory");
}

// ---------------------------------------------------------------------------
// Pass 2 (fast): WMMA correlation from pre-converted bf16, async LDS staging
// ---------------------------------------------------------------------------
__global__ __launch_bounds__(BLK) void corr_wmma_fast(
    const __bf16* __restrict__ f1w, const __bf16* __restrict__ f2w,
    float* __restrict__ out) {
  extern __shared__ __bf16 smem[];
  __bf16* f1s = smem;                       // [192][PITCH]
  __bf16* f2s = smem + F1_ELEMS;            // [288][PITCH], halo rows zeroed
  __bf16* f2i = f2s + 32 * PITCH;           // interior origin (x2 = 0)
  float*  pout = (float*)(smem + F1_ELEMS + F2_ELEMS);  // [21][PITCH2] + trash

  const int y    = blockIdx.x;
  const int b    = blockIdx.y;
  const int tid  = threadIdx.x;
  const int lane = tid & 31;
  const int wv   = tid >> 5;
  const int t0   = (wv >> 1) * 16;
  const int par  = wv & 1;

  // Async copy ownership: 16B chunk c16, row set x = xb + 24k, k = 0..7.
  const int c16 = tid & 15;
  const int xb  = tid >> 4;                  // 0..23
  const uint32_t lane_goff = (uint32_t)(xb * (CC * 2) + c16 * 16);
  const uint32_t f1s_lane  = (uint32_t)(uintptr_t)f1s + (uint32_t)(xb * (PITCH * 2) + c16 * 16);
  const uint32_t f2i_lane  = (uint32_t)(uintptr_t)f2i + (uint32_t)(xb * (PITCH * 2) + c16 * 16);

  // ---- stage feat1 row y via async DMA ----
  {
    const __bf16* rowb = f1w + ((size_t)(b * HH + y) * WW) * CC;
#pragma unroll
    for (int k = 0; k < 8; ++k)
      async_copy16(f1s_lane + k * (24 * PITCH * 2), lane_goff + k * (24 * CC * 2), rowb);
  }
  // ---- zero the feat2 halo rows once, b128 ----
  for (int idx = tid; idx < 96 * (CC / 8); idx += BLK) {
    const int c8 = idx & 15, hr = idx >> 4;
    const int xh = (hr < 32) ? hr : (hr - 32 + 224);
    bf16x8 z = {};
    *(bf16x8*)(f2s + xh * PITCH + c8 * 8) = z;
  }
  async_wait0();

  const __bf16* f2base = f2w + (size_t)b * HH * WW * CC;

  for (int dxi = 0; dxi < NOFF; ++dxi) {
    const int  r       = y + 2 * dxi - 20;
    const bool inRange = (r >= 0) && (r < HH);      // uniform across block
    float* ob = out + ((size_t)(b * (NOFF * NOFF) + dxi * NOFF) * HH + y) * WW;

    if (inRange) {
      __syncthreads();   // prior iteration done reading f2s and pout
      const __bf16* rowb = f2base + ((size_t)r * WW) * CC;
#pragma unroll
      for (int k = 0; k < 8; ++k)
        async_copy16(f2i_lane + k * (24 * PITCH * 2), lane_goff + k * (24 * CC * 2), rowb);
      async_wait0();
      __syncthreads();

      v8f acc[3] = {};
      band_gemm(f1s, f2s, t0, par, lane, acc);
      band_scatter(pout, acc, t0, par, lane);
      __syncthreads();   // band complete in pout

      for (int idx = tid; idx < NOFF * 48; idx += BLK) {
        const int dyi = idx / 48, xq = idx - dyi * 48;
        *(f32x4*)(ob + (size_t)dyi * PLANE + xq * 4) =
            *(const f32x4*)(pout + dyi * PITCH2 + xq * 4);
      }
    } else {
      const f32x4 z = {};
      for (int idx = tid; idx < NOFF * 48; idx += BLK) {
        const int dyi = idx / 48, xq = idx - dyi * 48;
        *(f32x4*)(ob + (size_t)dyi * PLANE + xq * 4) = z;
      }
    }
  }
}

// ---------------------------------------------------------------------------
// Fallback (no workspace): self-contained f32 -> bf16 staging
// ---------------------------------------------------------------------------
__global__ __launch_bounds__(BLK) void corr_wmma_kernel(
    const float* __restrict__ feat1, const float* __restrict__ feat2,
    float* __restrict__ out) {
  extern __shared__ __bf16 smem[];
  __bf16* f1s = smem;
  __bf16* f2s = smem + F1_ELEMS;
  __bf16* f2i = f2s + 32 * PITCH;
  float*  pout = (float*)(smem + F1_ELEMS + F2_ELEMS);

  const int y    = blockIdx.x;
  const int b    = blockIdx.y;
  const int tid  = threadIdx.x;
  const int lane = tid & 31;
  const int wv   = tid >> 5;
  const int t0   = (wv >> 1) * 16;
  const int par  = wv & 1;
  const int cg0 = tid & 7;
  const int x0  = (tid >> 3) * 4;

  {
    const float* src = feat1 + (size_t)b * CC * PLANE + (size_t)y * WW;
#pragma unroll
    for (int it = 0; it < 4; ++it) {
      const int c0 = cg0 * 4 + it * 32;
      f32x4 v0 = *(const f32x4*)(src + (size_t)(c0 + 0) * PLANE + x0);
      f32x4 v1 = *(const f32x4*)(src + (size_t)(c0 + 1) * PLANE + x0);
      f32x4 v2 = *(const f32x4*)(src + (size_t)(c0 + 2) * PLANE + x0);
      f32x4 v3 = *(const f32x4*)(src + (size_t)(c0 + 3) * PLANE + x0);
#pragma unroll
      for (int xi = 0; xi < 4; ++xi) {
        bf16x4 p;
        p[0] = (__bf16)v0[xi]; p[1] = (__bf16)v1[xi];
        p[2] = (__bf16)v2[xi]; p[3] = (__bf16)v3[xi];
        *(bf16x4*)(f1s + (size_t)(x0 + xi) * PITCH + c0) = p;
      }
    }
  }
  for (int idx = tid; idx < 96 * (CC / 8); idx += BLK) {
    const int c8 = idx & 15, hr = idx >> 4;
    const int xh = (hr < 32) ? hr : (hr - 32 + 224);
    bf16x8 z = {};
    *(bf16x8*)(f2s + xh * PITCH + c8 * 8) = z;
  }

  const float* f2base = feat2 + (size_t)b * CC * PLANE;

  for (int dxi = 0; dxi < NOFF; ++dxi) {
    const int  r       = y + 2 * dxi - 20;
    const bool inRange = (r >= 0) && (r < HH);
    float* ob = out + ((size_t)(b * (NOFF * NOFF) + dxi * NOFF) * HH + y) * WW;

    if (inRange) {
      const float* src = f2base + (size_t)r * WW;
      f32x4 buf[4][4];
#pragma unroll
      for (int it = 0; it < 4; ++it) {
        const int c0 = cg0 * 4 + it * 32;
#pragma unroll
        for (int j = 0; j < 4; ++j)
          buf[it][j] = *(const f32x4*)(src + (size_t)(c0 + j) * PLANE + x0);
      }
      __syncthreads();
#pragma unroll
      for (int it = 0; it < 4; ++it) {
        const int c0 = cg0 * 4 + it * 32;
#pragma unroll
        for (int xi = 0; xi < 4; ++xi) {
          bf16x4 p;
          p[0] = (__bf16)buf[it][0][xi]; p[1] = (__bf16)buf[it][1][xi];
          p[2] = (__bf16)buf[it][2][xi]; p[3] = (__bf16)buf[it][3][xi];
          *(bf16x4*)(f2i + (size_t)(x0 + xi) * PITCH + c0) = p;
        }
      }
      __syncthreads();

      v8f acc[3] = {};
      band_gemm(f1s, f2s, t0, par, lane, acc);
      band_scatter(pout, acc, t0, par, lane);
      __syncthreads();

      for (int idx = tid; idx < NOFF * 48; idx += BLK) {
        const int dyi = idx / 48, xq = idx - dyi * 48;
        *(f32x4*)(ob + (size_t)dyi * PLANE + xq * 4) =
            *(const f32x4*)(pout + dyi * PITCH2 + xq * 4);
      }
    } else {
      const f32x4 z = {};
      for (int idx = tid; idx < NOFF * 48; idx += BLK) {
        const int dyi = idx / 48, xq = idx - dyi * 48;
        *(f32x4*)(ob + (size_t)dyi * PLANE + xq * 4) = z;
      }
    }
  }
}

extern "C" void kernel_launch(void* const* d_in, const int* in_sizes, int n_in,
                              void* d_out, int out_size, void* d_ws, size_t ws_size,
                              hipStream_t stream) {
  const float* feat1 = (const float*)d_in[0];
  const float* feat2 = (const float*)d_in[1];
  float* out = (float*)d_out;
  (void)in_sizes; (void)n_in; (void)out_size;

  if (ws_size >= WS_NEED) {
    __bf16* w = (__bf16*)d_ws;
    cvt_kernel<<<dim3(HH, 4, 2), dim3(BLK), 0, stream>>>(feat1, feat2, w);
    corr_wmma_fast<<<dim3(HH, 4), dim3(BLK), LDS_SZ, stream>>>(
        w, w + TENSOR_ELEMS, out);
  } else {
    corr_wmma_kernel<<<dim3(HH, 4), dim3(BLK), LDS_SZ, stream>>>(
        feat1, feat2, out);
  }
}
